// DSASpikingTransformer_2130303779105
// MI455X (gfx1250) — compile-verified, matrix operand order checked by reading, not run on hardware
//
#include <hip/hip_runtime.h>
#include <cstdint>
#include <cstddef>

typedef __attribute__((ext_vector_type(16))) __bf16          v16bf;
typedef __attribute__((ext_vector_type(8)))  float           v8f;
typedef __attribute__((ext_vector_type(8)))  unsigned short  v8us;
typedef __attribute__((ext_vector_type(4)))  int             v4i;

// ---- CDNA5 async global->LDS path (ASYNCcnt), with safe fallback ----------
#if defined(__has_builtin)
#if __has_builtin(__builtin_amdgcn_global_load_async_to_lds_b128) && \
    __has_builtin(__builtin_amdgcn_s_wait_asynccnt)
#define USE_ASYNC_LDS 1
#endif
#endif
#ifndef USE_ASYNC_LDS
#define USE_ASYNC_LDS 0
#endif

#if USE_ASYNC_LDS
__device__ __forceinline__ void async_cp16(void* dst_lds, const void* src_glob) {
  __builtin_amdgcn_global_load_async_to_lds_b128(
      (__attribute__((address_space(1))) v4i*)src_glob,
      (__attribute__((address_space(3))) v4i*)dst_lds, 0, 0);
}
#endif

__device__ __forceinline__ unsigned short f2bf(float f) {
  unsigned int u = __float_as_uint(f);
  u += 0x7FFFu + ((u >> 16) & 1u);          // round-to-nearest-even
  return (unsigned short)(u >> 16);
}
__device__ __forceinline__ float bf2f(unsigned short h) {
  return __uint_as_float(((unsigned int)h) << 16);
}

enum { MODE_F32 = 0, MODE_BF16 = 1, MODE_SPIKE_F32 = 2, MODE_SPIKE_BF16 = 3 };

// ---------------------------------------------------------------------------
// fp32 -> bf16 elementwise
// ---------------------------------------------------------------------------
__global__ void cvt_kernel(const float* __restrict__ in,
                           unsigned short* __restrict__ o, size_t n) {
  for (size_t i = (size_t)blockIdx.x * blockDim.x + threadIdx.x; i < n;
       i += (size_t)gridDim.x * blockDim.x)
    o[i] = f2bf(in[i]);
}

// ---------------------------------------------------------------------------
// C[M,N] = epilogue( A[M,K](bf16) x W[N,K]^T(bf16) + bias[N] )
// Block tile 128(M) x 64(N), 8 waves; each wave owns a 16x64 strip.
// K stepped by 32 -> one v_wmma_f32_16x16x32_bf16 per 16x16 subtile per step.
// Staging: async global->LDS double buffer (ASYNCcnt) or register pipeline.
// ---------------------------------------------------------------------------
__global__ __launch_bounds__(256) void gemm_bf16_kernel(
    const unsigned short* __restrict__ A,
    const unsigned short* __restrict__ W,
    const float* __restrict__ bias,
    float* __restrict__ Cf,
    unsigned short* __restrict__ Cb,
    int Mdim, int Ndim, int Kdim, int mode)
{
#if USE_ASYNC_LDS
  __shared__ unsigned short As[2][128 * 32];
  __shared__ unsigned short Ws[2][64 * 32];
#else
  __shared__ unsigned short As[1][128 * 32];
  __shared__ unsigned short Ws[1][64 * 32];
#endif
  const int tid  = threadIdx.x;
  const int wave = tid >> 5;
  const int lane = tid & 31;
  const int hf   = lane >> 4;     // K-half selector for 16-bit operand layout
  const int lr   = lane & 15;     // row (A) / column (B) index
  const int mBase = blockIdx.y * 128;
  const int nBase = blockIdx.x * 64;
  const int i0 = tid * 8;         // this thread's 16B slot(s) in the tile
  const int i1 = i0 + 2048;
  const size_t ga0 = (size_t)(mBase + (i0 >> 5)) * Kdim + (i0 & 31);
  const size_t ga1 = (size_t)(mBase + (i1 >> 5)) * Kdim + (i1 & 31);
  const size_t gw0 = (size_t)(nBase + (i0 >> 5)) * Kdim + (i0 & 31);

  v8f acc[4] = {};
  const int nK = Kdim >> 5;

#if USE_ASYNC_LDS
  // ---- double-buffered LDS; memory->LDS without touching VGPRs ----
  async_cp16(&As[0][i0], &A[ga0]);
  async_cp16(&As[0][i1], &A[ga1]);
  async_cp16(&Ws[0][i0], &W[gw0]);
  for (int t = 0; t < nK; ++t) {
    const int cur = t & 1;
    if (t + 1 < nK) {
      const int nxt = cur ^ 1;
      const int k0 = (t + 1) << 5;
      async_cp16(&As[nxt][i0], &A[ga0 + k0]);
      async_cp16(&As[nxt][i1], &A[ga1 + k0]);
      async_cp16(&Ws[nxt][i0], &W[gw0 + k0]);
      __builtin_amdgcn_s_wait_asynccnt(3);   // tile t retired (in-order)
    } else {
      __builtin_amdgcn_s_wait_asynccnt(0);
    }
    __syncthreads();                         // tile t visible to all waves
    v16bf afrag = *(const v16bf*)&As[cur][(wave * 16 + lr) * 32 + hf * 16];
    #pragma unroll
    for (int nt = 0; nt < 4; ++nt) {
      v16bf bfrag = *(const v16bf*)&Ws[cur][(nt * 16 + lr) * 32 + hf * 16];
      acc[nt] = __builtin_amdgcn_wmma_f32_16x16x32_bf16(
          false, afrag, false, bfrag, (short)0, acc[nt], false, false);
    }
    __syncthreads();                         // all reads done before overwrite
  }
#else
  // ---- register pipeline: next tile's global loads overlap the WMMAs ----
  v8us ra0 = *(const v8us*)&A[ga0];
  v8us ra1 = *(const v8us*)&A[ga1];
  v8us rw0 = *(const v8us*)&W[gw0];
  for (int t = 0; t < nK; ++t) {
    __syncthreads();
    *(v8us*)&As[0][i0] = ra0;
    *(v8us*)&As[0][i1] = ra1;
    *(v8us*)&Ws[0][i0] = rw0;
    __syncthreads();
    if (t + 1 < nK) {
      const int k0 = (t + 1) << 5;
      ra0 = *(const v8us*)&A[ga0 + k0];      // in flight during WMMAs below
      ra1 = *(const v8us*)&A[ga1 + k0];
      rw0 = *(const v8us*)&W[gw0 + k0];
    }
    v16bf afrag = *(const v16bf*)&As[0][(wave * 16 + lr) * 32 + hf * 16];
    #pragma unroll
    for (int nt = 0; nt < 4; ++nt) {
      v16bf bfrag = *(const v16bf*)&Ws[0][(nt * 16 + lr) * 32 + hf * 16];
      acc[nt] = __builtin_amdgcn_wmma_f32_16x16x32_bf16(
          false, afrag, false, bfrag, (short)0, acc[nt], false, false);
    }
  }
#endif

  // C layout: VGPR r, lanes<16 -> M=r ; lanes>=16 -> M=8+r ; N = lane&15
  #pragma unroll
  for (int nt = 0; nt < 4; ++nt) {
    int n = nBase + nt * 16 + lr;
    float bvv = bias ? bias[n] : 0.0f;
    #pragma unroll
    for (int r = 0; r < 8; ++r) {
      int m = mBase + wave * 16 + hf * 8 + r;
      float v = acc[nt][r] + bvv;
      if (mode >= MODE_SPIKE_F32) v = (v > 0.5f) ? 1.0f : 0.0f;   // LIF spike
      size_t off = (size_t)m * Ndim + n;
      if (mode == MODE_F32 || mode == MODE_SPIKE_F32) Cf[off] = v;
      else                                            Cb[off] = f2bf(v);
    }
  }
}

// ---------------------------------------------------------------------------
// Dynamic sparse attention: 8 waves per (batch, head, 16-query tile).
// Phase 1: waves split key tiles 8-way, WMMA scores -> 128KB LDS row buffer.
// Phase 2: waves split the 16 query rows (2 each): exact top-k via iterative
//          wave argmax, softmax over selected, sparse p.V gather.
// ---------------------------------------------------------------------------
__global__ __launch_bounds__(256) void dsa_attn_kernel(
    const unsigned short* __restrict__ Q,
    const unsigned short* __restrict__ Kmat,
    const unsigned short* __restrict__ V,
    float* __restrict__ Omat,
    const int* __restrict__ tkp, int T)
{
  extern __shared__ float sc[];                 // [16][T] fp32 scores
  const int D = 512, dh = 64;
  const int qt = blockIdx.x;
  const int hd = blockIdx.y;
  const int b  = blockIdx.z;
  const int tid  = threadIdx.x;
  const int wave = tid >> 5;
  const int lane = tid & 31;
  const int hf = lane >> 4, lr = lane & 15;
  const size_t base = (size_t)b * T * D;
  const size_t hoff = (size_t)hd * dh;

  const unsigned short* qrow = Q + base + (size_t)(qt * 16 + lr) * D + hoff;
  v16bf a0 = *(const v16bf*)(qrow + hf * 16);        // d = 0..31 chunk
  v16bf a1 = *(const v16bf*)(qrow + 32 + hf * 16);   // d = 32..63 chunk

  const float scale = 0.125f;                        // 1/sqrt(64)
  for (int kt = wave; kt < T / 16; kt += 8) {        // uniform: (T/16)%8 == 0
    const unsigned short* krow = Kmat + base + (size_t)(kt * 16 + lr) * D + hoff;
    v16bf b0 = *(const v16bf*)(krow + hf * 16);
    v16bf b1 = *(const v16bf*)(krow + 32 + hf * 16);
    v8f cacc = {};
    cacc = __builtin_amdgcn_wmma_f32_16x16x32_bf16(false, a0, false, b0, (short)0, cacc, false, false);
    cacc = __builtin_amdgcn_wmma_f32_16x16x32_bf16(false, a1, false, b1, (short)0, cacc, false, false);
    #pragma unroll
    for (int rr = 0; rr < 8; ++rr)
      sc[(size_t)(hf * 8 + rr) * T + kt * 16 + lr] = cacc[rr] * scale;
  }
  __syncthreads();

  int topk = *tkp;                                   // device scalar (graph-safe)
  if (topk < 1)  topk = 1;
  if (topk > 32) topk = 32;

  for (int row = wave * 2; row < wave * 2 + 2; ++row) {
    volatile float* s = &sc[(size_t)row * T];
    float selv = 0.0f; int seli = 0;                 // lane `it` keeps it-th max
    for (int it = 0; it < topk; ++it) {
      float bvv = -3.0e38f; int bi = 0;
      for (int j = lane; j < T; j += 32) {
        float v = s[j];
        if (v > bvv) { bvv = v; bi = j; }
      }
      #pragma unroll
      for (int off = 16; off > 0; off >>= 1) {       // argmax, smaller-index ties
        float ov = __shfl_xor(bvv, off, 32);
        int   oi = __shfl_xor(bi,  off, 32);
        if (ov > bvv || (ov == bvv && oi < bi)) { bvv = ov; bi = oi; }
      }
      if (lane == it) { selv = bvv; seli = bi; }
      if (lane == 0) s[bi] = -3.0e38f;               // mask winner
    }
    float mx  = __shfl(selv, 0, 32);                 // global max = first pick
    float pr  = (lane < topk) ? __expf(selv - mx) : 0.0f;
    float tot = pr;
    #pragma unroll
    for (int off = 16; off > 0; off >>= 1) tot += __shfl_xor(tot, off, 32);
    pr /= tot;

    float o0 = 0.0f, o1 = 0.0f;                      // each lane: dims 2l, 2l+1
    for (int j = 0; j < topk; ++j) {
      float pj = __shfl(pr, j, 32);
      int   ij = __shfl(seli, j, 32);
      const unsigned short* vr = V + base + (size_t)ij * D + hoff + 2 * lane;
      o0 += pj * bf2f(vr[0]);
      o1 += pj * bf2f(vr[1]);
    }
    float* orow = Omat + base + (size_t)(qt * 16 + row) * D + hoff + 2 * lane;
    orow[0] = o0; orow[1] = o1;
  }
}

// ---------------------------------------------------------------------------
// y = LayerNorm(x + r) * g + b  (D = 512, one 256-thread block per row)
// also emits bf16 copy for the next WMMA GEMM
// ---------------------------------------------------------------------------
__global__ __launch_bounds__(256) void resln_kernel(
    const float* __restrict__ x, const float* __restrict__ r,
    const float* __restrict__ g, const float* __restrict__ bb,
    float* __restrict__ y, unsigned short* __restrict__ yb)
{
  const int D = 512;
  int row = blockIdx.x;
  int tid = threadIdx.x;
  const float* xr = x + (size_t)row * D;
  float v0 = xr[tid];
  float v1 = xr[tid + 256];
  if (r) { v0 += r[(size_t)row * D + tid]; v1 += r[(size_t)row * D + tid + 256]; }
  float s = v0 + v1, sq = v0 * v0 + v1 * v1;
  #pragma unroll
  for (int off = 16; off > 0; off >>= 1) {
    s  += __shfl_xor(s,  off, 32);
    sq += __shfl_xor(sq, off, 32);
  }
  __shared__ float sh[16];
  int w = tid >> 5;
  if ((tid & 31) == 0) { sh[w] = s; sh[8 + w] = sq; }
  __syncthreads();
  if (tid == 0) {
    float ts = 0, tq = 0;
    for (int i = 0; i < 8; ++i) { ts += sh[i]; tq += sh[8 + i]; }
    sh[0] = ts; sh[1] = tq;
  }
  __syncthreads();
  float mean = sh[0] / D;
  float var  = sh[1] / D - mean * mean;
  float rstd = rsqrtf(var + 1e-5f);
  float y0 = (v0 - mean) * rstd * g[tid]       + bb[tid];
  float y1 = (v1 - mean) * rstd * g[tid + 256] + bb[tid + 256];
  float* yr = y + (size_t)row * D;
  yr[tid] = y0; yr[tid + 256] = y1;
  if (yb) {
    yb[(size_t)row * D + tid]       = f2bf(y0);
    yb[(size_t)row * D + tid + 256] = f2bf(y1);
  }
}

// ---------------------------------------------------------------------------
// misc small kernels
// ---------------------------------------------------------------------------
__global__ void addpos_kernel(float* __restrict__ h, const float* __restrict__ pos,
                              int Bn, int TD) {
  size_t n = (size_t)Bn * TD;
  for (size_t i = (size_t)blockIdx.x * blockDim.x + threadIdx.x; i < n;
       i += (size_t)gridDim.x * blockDim.x)
    h[i] += pos[i % TD];
}

__global__ void pool_kernel(const float* __restrict__ h, float* __restrict__ pooled,
                            int T, int D) {
  int d = blockIdx.x * blockDim.x + threadIdx.x;
  int b = blockIdx.y;
  if (d >= D) return;
  const float* p = h + (size_t)b * T * D + d;
  float s = 0.0f;
  for (int t = 0; t < T; ++t) s += p[(size_t)t * D];
  pooled[(size_t)b * D + d] = s / (float)T;
}

__global__ void cls_kernel(const float* __restrict__ pooled, const float* __restrict__ w,
                           const float* __restrict__ bias, float* __restrict__ out,
                           int D, int O) {
  int o = blockIdx.x * blockDim.x + threadIdx.x;
  int b = blockIdx.y;
  if (o >= O) return;
  float s = bias[o];
  const float* pp = pooled + (size_t)b * D;
  const float* wr = w + (size_t)o * D;
  for (int d = 0; d < D; ++d) s += pp[d] * wr[d];
  out[(size_t)b * O + o] = s;
}

// ---------------------------------------------------------------------------
extern "C" void kernel_launch(void* const* d_in, const int* in_sizes, int n_in,
                              void* d_out, int out_size, void* d_ws, size_t ws_size,
                              hipStream_t stream) {
  const float* x     = (const float*)d_in[0];
  const float* emb_w = (const float*)d_in[1];
  const float* emb_b = (const float*)d_in[2];
  const float* pos   = (const float*)d_in[3];
  const float* wq    = (const float*)d_in[4];
  const float* bq    = (const float*)d_in[5];
  const float* wk    = (const float*)d_in[6];
  const float* bk    = (const float*)d_in[7];
  const float* wv    = (const float*)d_in[8];
  const float* bv    = (const float*)d_in[9];
  const float* wo    = (const float*)d_in[10];
  const float* bo    = (const float*)d_in[11];
  const float* ln1g  = (const float*)d_in[12];
  const float* ln1b  = (const float*)d_in[13];
  const float* f1w   = (const float*)d_in[14];
  const float* f1b   = (const float*)d_in[15];
  const float* f2w   = (const float*)d_in[16];
  const float* f2b   = (const float*)d_in[17];
  const float* ln2g  = (const float*)d_in[18];
  const float* ln2b  = (const float*)d_in[19];
  const float* fng   = (const float*)d_in[20];
  const float* fnb   = (const float*)d_in[21];
  const float* clsw  = (const float*)d_in[22];
  const float* clsb  = (const float*)d_in[23];
  const int*   tkp   = (const int*)d_in[24];
  float* out = (float*)d_out;
  (void)in_sizes; (void)n_in; (void)out_size; (void)ws_size;

  const int B = 2, T = 2048, D = 512, F = 2048, IN = 128, L = 4, O = 256;
  const int M = B * T;

  // ---- workspace carve (~93 MB total) ----
  char* p = (char*)d_ws;
  auto carve = [&](size_t bytes) { char* q = p; p += (bytes + 255) & ~((size_t)255); return q; };
  unsigned short* xb    = (unsigned short*)carve((size_t)M * IN * 2);
  unsigned short* embwb = (unsigned short*)carve((size_t)D * IN * 2);
  unsigned short* wqb   = (unsigned short*)carve((size_t)L * D * D * 2);
  unsigned short* wkb   = (unsigned short*)carve((size_t)L * D * D * 2);
  unsigned short* wvb   = (unsigned short*)carve((size_t)L * D * D * 2);
  unsigned short* wob   = (unsigned short*)carve((size_t)L * D * D * 2);
  unsigned short* f1wb  = (unsigned short*)carve((size_t)L * F * D * 2);
  unsigned short* f2wb  = (unsigned short*)carve((size_t)L * D * F * 2);
  float*          hbuf  = (float*)carve((size_t)M * D * 4);
  unsigned short* hb    = (unsigned short*)carve((size_t)M * D * 2);
  unsigned short* qb    = (unsigned short*)carve((size_t)M * D * 2);
  unsigned short* kb    = (unsigned short*)carve((size_t)M * D * 2);
  unsigned short* vb    = (unsigned short*)carve((size_t)M * D * 2);
  float*          attno = (float*)carve((size_t)M * D * 4);
  unsigned short* attnb = (unsigned short*)carve((size_t)M * D * 2);
  float*          asp   = (float*)carve((size_t)M * D * 4);
  unsigned short* hidb  = (unsigned short*)carve((size_t)M * F * 2);
  float*          ffo   = (float*)carve((size_t)M * D * 4);
  float*          pooled= (float*)carve((size_t)B * D * 4);

  // allow 128 KB dynamic LDS for the attention kernel (WGP has 320 KB)
  (void)hipFuncSetAttribute((const void*)dsa_attn_kernel,
                            hipFuncAttributeMaxDynamicSharedMemorySize,
                            16 * T * (int)sizeof(float));

  dim3 cb(256);
  auto cvt = [&](const float* src, unsigned short* dst, size_t n) {
    int g = (int)((n + 2047) / 2048); if (g < 1) g = 1; if (g > 4096) g = 4096;
    cvt_kernel<<<g, cb, 0, stream>>>(src, dst, n);
  };
  auto gemm = [&](const unsigned short* A, const unsigned short* W, const float* bias,
                  float* Cf, unsigned short* Cb, int m, int n, int k, int mode) {
    gemm_bf16_kernel<<<dim3(n / 64, m / 128), dim3(256), 0, stream>>>(
        A, W, bias, Cf, Cb, m, n, k, mode);
  };

  // weights -> bf16 (every call; deterministic)
  cvt(x,     xb,    (size_t)M * IN);
  cvt(emb_w, embwb, (size_t)D * IN);
  cvt(wq,    wqb,   (size_t)L * D * D);
  cvt(wk,    wkb,   (size_t)L * D * D);
  cvt(wv,    wvb,   (size_t)L * D * D);
  cvt(wo,    wob,   (size_t)L * D * D);
  cvt(f1w,   f1wb,  (size_t)L * F * D);
  cvt(f2w,   f2wb,  (size_t)L * D * F);

  // embedding + positional
  gemm(xb, embwb, emb_b, hbuf, nullptr, M, D, IN, MODE_F32);
  addpos_kernel<<<1024, cb, 0, stream>>>(hbuf, pos, B, T * D);
  cvt(hbuf, hb, (size_t)M * D);

  for (int l = 0; l < L; ++l) {
    size_t wdd = (size_t)l * D * D;
    gemm(hb, wqb + wdd, bq + (size_t)l * D, nullptr, qb, M, D, D, MODE_BF16);
    gemm(hb, wkb + wdd, bk + (size_t)l * D, nullptr, kb, M, D, D, MODE_BF16);
    gemm(hb, wvb + wdd, bv + (size_t)l * D, nullptr, vb, M, D, D, MODE_BF16);

    dsa_attn_kernel<<<dim3(T / 16, 8, B), dim3(256),
                      (size_t)16 * T * sizeof(float), stream>>>(qb, kb, vb, attno, tkp, T);

    cvt(attno, attnb, (size_t)M * D);
    gemm(attnb, wob + wdd, bo + (size_t)l * D, asp, nullptr, M, D, D, MODE_SPIKE_F32);
    resln_kernel<<<M, cb, 0, stream>>>(hbuf, asp,
        ln1g + (size_t)l * D, ln1b + (size_t)l * D, hbuf, hb);

    gemm(hb,   f1wb + (size_t)l * F * D, f1b + (size_t)l * F, nullptr, hidb, M, F, D, MODE_SPIKE_BF16);
    gemm(hidb, f2wb + (size_t)l * D * F, f2b + (size_t)l * D, ffo, nullptr, M, D, F, MODE_SPIKE_F32);
    resln_kernel<<<M, cb, 0, stream>>>(hbuf, ffo,
        ln2g + (size_t)l * D, ln2b + (size_t)l * D, hbuf, hb);
  }

  // final LN -> mean pool over T -> classifier
  resln_kernel<<<M, cb, 0, stream>>>(hbuf, nullptr, fng, fnb, hbuf, nullptr);
  pool_kernel<<<dim3(D / 256, B), cb, 0, stream>>>(hbuf, pooled, T, D);
  cls_kernel<<<dim3(1, B), cb, 0, stream>>>(pooled, clsw, clsb, out, D, O);
}
